// Model_26070451487074
// MI455X (gfx1250) — matrix operand order repaired and checked by reading, not compiled, not run on hardware
//
#include <hip/hip_runtime.h>
#include <cstdint>
#include <cstddef>

#define N_NODES  10000
#define N_EDGES  160000
#define IN_CH    768
#define TYPE_DIM 128
#define D1       896      // IN_CH + TYPE_DIM
#define HID      768
#define OUTD     768
#define NREL     3

typedef __attribute__((ext_vector_type(16))) __bf16 v16bf;
typedef __attribute__((ext_vector_type(8)))  __bf16 v8bf;
typedef __attribute__((ext_vector_type(8)))  float  v8f;
typedef __attribute__((ext_vector_type(4)))  int    v4i;

// Async global->LDS path (CDNA5): builtin confirmed present; params are
// (global int4*, shared int4*, imm offset, imm cpol) per the round-2 diagnostic.
#if defined(__HIP_DEVICE_COMPILE__) && __has_builtin(__builtin_amdgcn_global_load_async_to_lds_b128)
#define HAVE_ASYNC_LDS 1
#define AS1V4(p) ((__attribute__((address_space(1))) v4i*)(p))
#define AS3V4(p) ((__attribute__((address_space(3))) v4i*)(p))
#else
#define HAVE_ASYNC_LDS 0
#endif

__device__ __forceinline__ void wait_async_lds() {
#if HAVE_ASYNC_LDS
#if __has_builtin(__builtin_amdgcn_s_wait_asynccnt)
    __builtin_amdgcn_s_wait_asynccnt(0);
#else
    asm volatile("s_wait_asynccnt 0x0" ::: "memory");
#endif
#endif
}

// ---------------- h0 = concat(x, node_emb[node_type]) ----------------
__global__ void concat_kernel(const float* __restrict__ x, const int* __restrict__ nt,
                              const float* __restrict__ emb, float* __restrict__ h) {
    long idx = (long)blockIdx.x * blockDim.x + threadIdx.x;
    if (idx >= (long)N_NODES * D1) return;
    int n = (int)(idx / D1), k = (int)(idx % D1);
    float v = (k < IN_CH) ? x[(size_t)n * IN_CH + k]
                          : emb[(size_t)nt[n] * TYPE_DIM + (k - IN_CH)];
    h[idx] = v;
}

// ---------------- per-(dst, rel) incoming-edge counts ----------------
__global__ void count_kernel(const int* __restrict__ dst, const int* __restrict__ et,
                             int* __restrict__ cnt) {
    int e = blockIdx.x * blockDim.x + threadIdx.x;
    if (e >= N_EDGES) return;
    atomicAdd(&cnt[dst[e] * NREL + et[e]], 1);
}

__global__ void inv_kernel(const int* __restrict__ cnt, float* __restrict__ inv) {
    int i = blockIdx.x * blockDim.x + threadIdx.x;
    if (i >= N_NODES * NREL) return;
    int c = cnt[i];
    inv[i] = (c > 0) ? 1.0f / (float)c : 0.0f;
}

// ---------------- edge scatter: agg[dst, rel, :] += h[src, :] * inv ----------------
__global__ __launch_bounds__(128) void agg_kernel(const float* __restrict__ h,
        const int* __restrict__ src, const int* __restrict__ dst,
        const int* __restrict__ et, const float* __restrict__ inv,
        float* __restrict__ agg, int d) {
    int e = blockIdx.x;
    int s = src[e], t = dst[e], r = et[e];
    float scale = inv[t * NREL + r];
    const float* sp = h + (size_t)s * d;
    float* dp = agg + ((size_t)t * NREL + r) * d;
    for (int k = threadIdx.x; k < d; k += 128)
        atomicAdd(&dp[k], sp[k] * scale);
}

// ---------------- pack A = [agg_r0|agg_r1|agg_r2|h] to bf16 ----------------
__global__ void pack_kernel(const float* __restrict__ agg, const float* __restrict__ h,
                            __bf16* __restrict__ A, int d, int K4, long total) {
    long idx = (long)blockIdx.x * blockDim.x + threadIdx.x;
    if (idx >= total) return;
    int n = (int)(idx / K4), k = (int)(idx % K4);
    float v = (k < NREL * d) ? agg[(size_t)n * NREL * d + k]
                             : h[(size_t)n * d + (k - NREL * d)];
    A[idx] = (__bf16)v;
}

// ---------------- WT[out_col][k] = stacked [w_rel; w_root] transposed, bf16 ----------------
__global__ void wconv_kernel(const float* __restrict__ wrel, const float* __restrict__ wroot,
                             __bf16* __restrict__ WT, int d, int e, long total) {
    long idx = (long)blockIdx.x * blockDim.x + threadIdx.x;
    if (idx >= total) return;
    int K4 = (NREL + 1) * d;
    int j = (int)(idx / K4), k = (int)(idx % K4);
    // w_rel is [R, d, e] contiguous -> element (r*d+row, j) at wrel[k*e + j]
    float v = (k < NREL * d) ? wrel[(size_t)k * e + j]
                             : wroot[(size_t)(k - NREL * d) * e + j];
    WT[idx] = (__bf16)v;
}

// ---------------- fused GEMM: out = relu(A[M,K] @ WT^T + bias), bf16 WMMA ----------------
// 256 threads = 8 waves; block tile 128x64; wave w owns rows [w*16, w*16+16) x all 64 cols
// (4 accumulators, A fragment reused 4x). Double-buffered LDS, one barrier per k-step.
#define LDA 40   // LDS row stride in bf16 (80 B, keeps 16B-aligned chunks, spreads banks)

__device__ __forceinline__ void load_tiles(const __bf16* __restrict__ A,
                                           const __bf16* __restrict__ WT,
                                           __bf16* as, __bf16* bs,
                                           int rbase, int cbase, int M, int K,
                                           int k0, int tid) {
    const int row = tid >> 2;          // 0..63
    const int ch  = (tid & 3) * 8;     // 0,8,16,24
    int gr0 = rbase + row;        if (gr0 >= M) gr0 = M - 1;   // clamp: C rows >= M never stored
    int gr1 = rbase + 64 + row;   if (gr1 >= M) gr1 = M - 1;
    const __bf16* g0 = A  + (size_t)gr0 * K + k0 + ch;
    const __bf16* g1 = A  + (size_t)gr1 * K + k0 + ch;
    const __bf16* gb = WT + (size_t)(cbase + row) * K + k0 + ch;
    __bf16* l0 = as + row * LDA + ch;
    __bf16* l1 = as + (64 + row) * LDA + ch;
    __bf16* lb = bs + row * LDA + ch;
#if HAVE_ASYNC_LDS
    __builtin_amdgcn_global_load_async_to_lds_b128(AS1V4((void*)g0), AS3V4((void*)l0), 0, 0);
    __builtin_amdgcn_global_load_async_to_lds_b128(AS1V4((void*)g1), AS3V4((void*)l1), 0, 0);
    __builtin_amdgcn_global_load_async_to_lds_b128(AS1V4((void*)gb), AS3V4((void*)lb), 0, 0);
#else
    *(v8bf*)l0 = *(const v8bf*)g0;
    *(v8bf*)l1 = *(const v8bf*)g1;
    *(v8bf*)lb = *(const v8bf*)gb;
#endif
}

__global__ __launch_bounds__(256) void rgcn_gemm_bf16(
        const __bf16* __restrict__ A,    // [M, K]
        const __bf16* __restrict__ WT,   // [Nout, K] (transposed weights)
        const float*  __restrict__ bias, // [Nout]
        float*        __restrict__ out,  // [M, Nout]
        int M, int K, int Nout) {
    __shared__ __bf16 As[2][128 * LDA];
    __shared__ __bf16 Bs[2][64 * LDA];

    const int tid  = threadIdx.x;
    const int wave = tid >> 5;
    const int lane = tid & 31;
    const int half = lane >> 4;
    const int ln   = lane & 15;
    const int wm   = wave * 16;          // wave row offset in 128-row block tile
    const int rbase = blockIdx.x * 128;
    const int cbase = blockIdx.y * 64;

    v8f acc[4] = {{}, {}, {}, {}};

    // preload k-tile 0 into buffer 0
    load_tiles(A, WT, As[0], Bs[0], rbase, cbase, M, K, 0, tid);
    wait_async_lds();
    __syncthreads();

    int buf = 0;
    for (int k0 = 0; k0 < K; k0 += 32) {
        // kick off next k-tile into the other buffer (overlaps with WMMA below)
        if (k0 + 32 < K)
            load_tiles(A, WT, As[buf ^ 1], Bs[buf ^ 1], rbase, cbase, M, K, k0 + 32, tid);

        // A fragment (16x32): lane<16 holds K {0..7}U{16..23}, lane>=16 holds {8..15}U{24..31}
        const __bf16* ap = &As[buf][(wm + ln) * LDA + 8 * half];
        v8bf alo = *(const v8bf*)ap;
        v8bf ahi = *(const v8bf*)(ap + 16);
        v16bf af;
#pragma unroll
        for (int i = 0; i < 8; ++i) { af[i] = alo[i]; af[i + 8] = ahi[i]; }

        // B fragments (32x16): lane<16 holds K {0..7}U{16..23}, lane>=16 holds {8..15}U{24..31}, N=lane%16
#pragma unroll
        for (int t = 0; t < 4; ++t) {
            const __bf16* bp = &Bs[buf][(t * 16 + ln) * LDA + 8 * half];
            v8bf blo = *(const v8bf*)bp;
            v8bf bhi = *(const v8bf*)(bp + 16);
            v16bf bfr;
#pragma unroll
            for (int i = 0; i < 8; ++i) { bfr[i] = blo[i]; bfr[i + 8] = bhi[i]; }
            acc[t] = __builtin_amdgcn_wmma_f32_16x16x32_bf16(false, af, false, bfr,
                                                             (short)0, acc[t], false, false);
        }

        wait_async_lds();     // our async loads into buf^1 done (no-op on fallback path)
        __syncthreads();      // everyone finished reading buf & writing buf^1
        buf ^= 1;
    }

    // epilogue: bias + relu; C/D layout: element i -> row i+8*half, col = lane%16
#pragma unroll
    for (int t = 0; t < 4; ++t) {
        int col = cbase + t * 16 + ln;
        float bb = bias[col];
#pragma unroll
        for (int i = 0; i < 8; ++i) {
            int gr = rbase + wm + i + 8 * half;
            if (gr < M) {
                float v = acc[t][i] + bb;
                out[(size_t)gr * Nout + col] = v > 0.f ? v : 0.f;
            }
        }
    }
}

// ---------------- driver ----------------
extern "C" void kernel_launch(void* const* d_in, const int* in_sizes, int n_in,
                              void* d_out, int out_size, void* d_ws, size_t ws_size,
                              hipStream_t stream) {
    const float* x         = (const float*)d_in[0];
    const int*   node_type = (const int*)d_in[1];
    const int*   edge_index= (const int*)d_in[2];
    const int*   edge_type = (const int*)d_in[3];
    const float* node_emb  = (const float*)d_in[4];
    const float* w_rel1 = (const float*)d_in[5];
    const float* w_root1= (const float*)d_in[6];
    const float* b1     = (const float*)d_in[7];
    const float* w_rel2 = (const float*)d_in[8];
    const float* w_root2= (const float*)d_in[9];
    const float* b2     = (const float*)d_in[10];
    const float* w_rel3 = (const float*)d_in[11];
    const float* w_root3= (const float*)d_in[12];
    const float* b3     = (const float*)d_in[13];

    const int* src = edge_index;
    const int* dst = edge_index + N_EDGES;

    uint8_t* p = (uint8_t*)d_ws;
    auto alloc = [&](size_t bytes) {
        uint8_t* r = p;
        p += (bytes + 255) & ~(size_t)255;
        return r;
    };
    float*  hA  = (float*)alloc((size_t)N_NODES * D1 * 4);
    float*  hB  = (float*)alloc((size_t)N_NODES * D1 * 4);
    float*  agg = (float*)alloc((size_t)N_NODES * NREL * D1 * 4);
    __bf16* Abf = (__bf16*)alloc((size_t)N_NODES * (NREL + 1) * D1 * 2);
    __bf16* WT  = (__bf16*)alloc((size_t)D1 * (NREL + 1) * D1 * 2);
    float*  inv = (float*)alloc((size_t)N_NODES * NREL * 4);
    int*    cnt = (int*)alloc((size_t)N_NODES * NREL * 4);

    // h0 = concat(x, node_emb[node_type])
    {
        long total = (long)N_NODES * D1;
        concat_kernel<<<(unsigned)((total + 255) / 256), 256, 0, stream>>>(x, node_type, node_emb, hA);
    }
    // per-(node, rel) mean normalizers (edge structure is layer-invariant -> compute once)
    (void)hipMemsetAsync(cnt, 0, (size_t)N_NODES * NREL * 4, stream);
    count_kernel<<<(N_EDGES + 255) / 256, 256, 0, stream>>>(dst, edge_type, cnt);
    inv_kernel<<<(N_NODES * NREL + 255) / 256, 256, 0, stream>>>(cnt, inv);

    auto layer = [&](const float* hin, float* hout, int d, int e,
                     const float* wrel, const float* wroot, const float* bias) {
        int K4 = (NREL + 1) * d;
        (void)hipMemsetAsync(agg, 0, (size_t)N_NODES * NREL * d * 4, stream);
        agg_kernel<<<N_EDGES, 128, 0, stream>>>(hin, src, dst, edge_type, inv, agg, d);
        long ptotal = (long)N_NODES * K4;
        pack_kernel<<<(unsigned)((ptotal + 255) / 256), 256, 0, stream>>>(agg, hin, Abf, d, K4, ptotal);
        long wtotal = (long)e * K4;
        wconv_kernel<<<(unsigned)((wtotal + 255) / 256), 256, 0, stream>>>(wrel, wroot, WT, d, e, wtotal);
        dim3 grid((N_NODES + 127) / 128, e / 64);
        rgcn_gemm_bf16<<<grid, 256, 0, stream>>>(Abf, WT, bias, hout, N_NODES, K4, e);
    };

    layer(hA, hB,            D1,  D1,   w_rel1, w_root1, b1);   // 896 -> 896
    layer(hB, hA,            D1,  HID,  w_rel2, w_root2, b2);   // 896 -> 768
    layer(hA, (float*)d_out, HID, OUTD, w_rel3, w_root3, b3);   // 768 -> 768
}
